// RWKV5TimeMix_13073880449278
// MI455X (gfx1250) — matrix operand order repaired and checked by reading, not compiled, not run on hardware
//
#include <hip/hip_runtime.h>
#include <stdint.h>

// ---------------------------------------------------------------------------
// RWKV5 TimeMix for MI455X (gfx1250): bf16 WMMA pipeline + async/TDM copies.
// B=4, L=2048, C=2048, H=32, K=64, TC=128, N=L/TC=16, M=B*L=8192.
// ---------------------------------------------------------------------------

#define Bsz 4
#define Lsz 2048
#define Csz 2048
#define Hsz 32
#define Ksz 64
#define TCsz 128
#define Msz (Bsz * Lsz)   // 8192

typedef __attribute__((ext_vector_type(16))) __bf16 bf16x16;
typedef __attribute__((ext_vector_type(8)))  float  f32x8;
typedef __attribute__((ext_vector_type(4)))  unsigned int u32x4;
typedef __attribute__((ext_vector_type(8)))  int i32x8;
typedef __attribute__((ext_vector_type(4)))  int i32x4;

union Frag {
    bf16x16 v;
    uint4   q[2];
};

__device__ __forceinline__ unsigned short f2bf(float x) {
    union { float f; uint32_t u; } v; v.f = x;
    uint32_t r = v.u + 0x7FFFu + ((v.u >> 16) & 1u);  // round-to-nearest-even
    return (unsigned short)(r >> 16);
}
__device__ __forceinline__ float bf2f(unsigned short s) {
    union { uint32_t u; float f; } v; v.u = ((uint32_t)s) << 16;
    return v.f;
}
__device__ __forceinline__ f32x8 wmma_bf16(bf16x16 a, bf16x16 b, f32x8 c) {
    return __builtin_amdgcn_wmma_f32_16x16x32_bf16(false, a, false, b, (short)0, c,
                                                   false, false);
}
// low 32 bits of a flat pointer to LDS == LDS byte offset (aperture rule)
__device__ __forceinline__ unsigned lds_off32(const void* p) {
    return (unsigned)(uintptr_t)p;
}

// ---------------------------------------------------------------------------
// K0: fp32 -> bf16 weight conversion
// ---------------------------------------------------------------------------
__global__ void __launch_bounds__(256) cvt_w(const float* __restrict__ src,
                                             unsigned short* __restrict__ dst) {
    size_t idx = (size_t)blockIdx.x * 256 + threadIdx.x;  // C*C grid
    dst[idx] = f2bf(src[idx]);
}

// ---------------------------------------------------------------------------
// K1: token shift + 4 lerps, emit bf16 activations
// ---------------------------------------------------------------------------
__global__ void __launch_bounds__(256) prep_lerp(
    const float* __restrict__ x, const float* __restrict__ shift,
    const float* __restrict__ tmk, const float* __restrict__ tmv,
    const float* __restrict__ tmr, const float* __restrict__ tmg,
    unsigned short* __restrict__ xk, unsigned short* __restrict__ xv,
    unsigned short* __restrict__ xr, unsigned short* __restrict__ xg) {
    size_t idx = (size_t)blockIdx.x * 256 + threadIdx.x;  // B*L*C
    int c = (int)(idx & (Csz - 1));
    size_t bl = idx >> 11;
    int l = (int)(bl & (Lsz - 1));
    int b = (int)(bl >> 11);
    float xc = x[idx];
    float xx = (l == 0) ? shift[(size_t)b * Csz + c] : x[idx - Csz];
    float mk = tmk[c], mv = tmv[c], mr = tmr[c], mg = tmg[c];
    xk[idx] = f2bf(xc * mk + xx * (1.f - mk));
    xv[idx] = f2bf(xc * mv + xx * (1.f - mv));
    xr[idx] = f2bf(xc * mr + xx * (1.f - mr));
    xg[idx] = f2bf(xc * mg + xx * (1.f - mg));
}

// ---------------------------------------------------------------------------
// K2/K5: bf16 NT GEMM  Y[M,N] = A[M,K] * Bw[N,K]^T   (K contiguous both sides)
// 128x128 block tile, BK=32, 256 threads = 8 waves (4x2), each wave 32x64.
// Tiles staged with GLOBAL_LOAD_ASYNC_TO_LDS_B128 (ASYNCcnt), double-buffered.
// MODE 0: bf16 output; MODE 1: f32 output.
// ---------------------------------------------------------------------------
template <int MODE>
__global__ void __launch_bounds__(256) gemm_nt(
    const unsigned short* __restrict__ A, const unsigned short* __restrict__ Bw,
    void* __restrict__ Y, int M, int N, int Kc) {
    __shared__ __align__(16) unsigned short As[2][128 * 32];
    __shared__ __align__(16) unsigned short Bs[2][128 * 32];
    const int tid = threadIdx.x;
    const int bm = blockIdx.y, bn = blockIdx.x;
    const int row  = tid >> 1;
    const int col0 = (tid & 1) << 4;
    const unsigned aByte = (unsigned)(((bm * 128 + row) * Kc + col0) * 2);
    const unsigned bByte = (unsigned)(((bn * 128 + row) * Kc + col0) * 2);
    const uint64_t pa = (uint64_t)(uintptr_t)A;
    const uint64_t pb = (uint64_t)(uintptr_t)Bw;
    const unsigned ldsA = lds_off32(&As[0][row * 32 + col0]);
    const unsigned ldsB = lds_off32(&Bs[0][row * 32 + col0]);
    const int KT = Kc >> 5;

    // async global->LDS tile copy: 32B per lane per matrix (2 x b128,
    // INST_OFFSET applies to both LDS and global address)
    auto issue = [&](int buf, int kt) {
        unsigned la = ldsA + buf * 8192;
        unsigned lb = ldsB + buf * 8192;
        unsigned va = aByte + kt * 64;
        unsigned vb = bByte + kt * 64;
        asm volatile(
            "global_load_async_to_lds_b128 %0, %1, %4\n\t"
            "global_load_async_to_lds_b128 %0, %1, %4 offset:16\n\t"
            "global_load_async_to_lds_b128 %2, %3, %5\n\t"
            "global_load_async_to_lds_b128 %2, %3, %5 offset:16"
            :: "v"(la), "v"(va), "v"(lb), "v"(vb), "s"(pa), "s"(pb)
            : "memory");
    };

    issue(0, 0);
    asm volatile("s_wait_asynccnt 0x0" ::: "memory");
    __syncthreads();

    const int wave = tid >> 5, lane = tid & 31;
    const int wm = (wave & 3) * 32, wn = (wave >> 2) * 64;
    const int lr = lane & 15, hi = lane >> 4;

    f32x8 acc[2][4];
#pragma unroll
    for (int i = 0; i < 2; ++i)
#pragma unroll
        for (int j = 0; j < 4; ++j) acc[i][j] = (f32x8)0.0f;

    for (int kt = 0; kt < KT; ++kt) {
        const int buf = kt & 1;
        const bool more = (kt + 1) < KT;
        if (more) issue(buf ^ 1, kt + 1);

        Frag af[2], bfr[4];
#pragma unroll
        for (int i = 0; i < 2; ++i) {
            int r = wm + i * 16 + lr;
            af[i].q[0] = *(const uint4*)&As[buf][r * 32 + hi * 8];
            af[i].q[1] = *(const uint4*)&As[buf][r * 32 + 16 + hi * 8];
        }
#pragma unroll
        for (int j = 0; j < 4; ++j) {
            int r = wn + j * 16 + lr;
            bfr[j].q[0] = *(const uint4*)&Bs[buf][r * 32 + hi * 8];
            bfr[j].q[1] = *(const uint4*)&Bs[buf][r * 32 + 16 + hi * 8];
        }
#pragma unroll
        for (int i = 0; i < 2; ++i)
#pragma unroll
            for (int j = 0; j < 4; ++j)
                acc[i][j] = wmma_bf16(af[i].v, bfr[j].v, acc[i][j]);

        if (more) {
            asm volatile("s_wait_asynccnt 0x0" ::: "memory");
            __syncthreads();
        }
    }
#pragma unroll
    for (int i = 0; i < 2; ++i)
#pragma unroll
        for (int j = 0; j < 4; ++j) {
            int m0 = bm * 128 + wm + i * 16 + hi * 8;
            int n0 = bn * 128 + wn + j * 16 + lr;
#pragma unroll
            for (int v = 0; v < 8; ++v) {
                float val = acc[i][j][v];
                size_t idx = (size_t)(m0 + v) * N + n0;
                if (MODE == 0) ((unsigned short*)Y)[idx] = f2bf(val);
                else           ((float*)Y)[idx] = val;
            }
        }
    (void)M;
}

// ---------------------------------------------------------------------------
// K3: chunked WKV attention. One workgroup per (b,h); 16 sequential chunks.
// Chunk tiles DMA'd by the Tensor Data Mover (TENSORcnt); all decays are
// integer powers of w (closed form), so every matmul is bf16 WMMA.
// ---------------------------------------------------------------------------
struct SmemK3 {
    unsigned short r_st[128 * 64];   // TDM staging: raw r chunk
    unsigned short k_st[128 * 64];   // TDM staging: raw k chunk
    unsigned short v_st[128 * 64];   // TDM staging: raw v chunk
    unsigned short rdw [128 * 64];   // r * w^(t-64), later r * w^t
    unsigned short kd  [128 * 64];   // k * w^(63-t)
    unsigned short kwT [64 * 128];   // (k * w^(127-t)) transposed [k][t]
    unsigned short vT  [64 * 128];   // v transposed [v][t]
    unsigned short a_bf[128 * 128];  // masked attention matrix (bf16)
    float          state[64 * 64];   // running kv state (f32)
    unsigned short stateT[64 * 64];  // state^T bf16 for WMMA B operand
    float          diag[128];
    float          wl[64];
    float          uu[64];
};

__global__ void __launch_bounds__(256) rwkv_att(
    const unsigned short* __restrict__ r_g, const unsigned short* __restrict__ k_g,
    const unsigned short* __restrict__ v_g, const float* __restrict__ tdecay,
    const float* __restrict__ tfaaaa, const float* __restrict__ wkv_in,
    float* __restrict__ att_out, float* __restrict__ state_out) {
    extern __shared__ __align__(16) char smraw[];
    SmemK3& sm = *(SmemK3*)smraw;
    const int tid = threadIdx.x;
    const int h = blockIdx.x & (Hsz - 1);
    const int b = blockIdx.x >> 5;

    if (tid < 64) {
        float w = __expf(-__expf(tdecay[h * Ksz + tid]));
        w = fmaxf(w, 0.005f);
        sm.wl[tid] = __logf(w);
        sm.uu[tid] = tfaaaa[h * Ksz + tid];
    }
    for (int i = tid; i < 4096; i += 256)
        sm.state[i] = wkv_in[(size_t)(b * Hsz + h) * 4096 + i];
    __syncthreads();

    const int wave = tid >> 5, lane = tid & 31;
    const int lr = lane & 15, hi = lane >> 4;

    // TDM descriptor group1 (same for r/k/v): 2-byte elements, 64x128 tile,
    // tensor (8192 x 2048), row stride 2048 elements.
    const i32x8 g1 = { (int)0x00010000u,          // data_size=2B, no flags
                       (int)(2048u << 16),        // tensor_dim0 = 2048 (lo16)
                       (int)(8192u << 16),        // tensor_dim1 = 8192 (lo16)
                       (int)(64u << 16),          // tile_dim0 = 64
                       128,                       // tile_dim1 = 128
                       2048,                      // tensor_dim0_stride lo32
                       0, 0 };
    const i32x4 z4 = { 0, 0, 0, 0 };
    const i32x8 z8 = { 0, 0, 0, 0, 0, 0, 0, 0 };

    for (int n = 0; n < 16; ++n) {
        const size_t gbase = ((size_t)(b * Lsz + n * TCsz) * Csz) + (size_t)h * Ksz;
        // ---- TDM: waves 0/1/2 DMA the r/k/v 128x64 bf16 tiles into LDS ----
        if (wave < 3) {
            const unsigned short* src = (wave == 0) ? r_g : (wave == 1) ? k_g : v_g;
            unsigned lo = (wave == 0) ? lds_off32(sm.r_st)
                        : (wave == 1) ? lds_off32(sm.k_st) : lds_off32(sm.v_st);
            uint64_t ga = (uint64_t)(uintptr_t)src + gbase * 2;
            u32x4 g0 = { 1u,                       // count=1 (valid user D#)
                         lo,                       // lds_addr
                         (unsigned)ga,             // global_addr[31:0]
                         (unsigned)(ga >> 32) | 0x80000000u };  // type=2
            __builtin_amdgcn_tensor_load_to_lds(g0, g1, z4, z4, z8, 0);
            __builtin_amdgcn_s_wait_tensorcnt((short)0);
        }
        __syncthreads();
        // ---- transform: scaled copies (powers of w) + v transpose ----
#pragma unroll 4
        for (int i = 0; i < 32; ++i) {
            int e = tid + i * 256;
            int row = e >> 6, col = e & 63;
            float rf = bf2f(sm.r_st[e]);
            sm.rdw[e] = f2bf(rf * __expf((float)(row - 64) * sm.wl[col]));
            float kf = bf2f(sm.k_st[e]);
            sm.kd[e] = f2bf(kf * __expf((float)(63 - row) * sm.wl[col]));
            sm.kwT[col * 128 + row] = f2bf(kf * __expf((float)(127 - row) * sm.wl[col]));
            sm.vT[col * 128 + row] = sm.v_st[e];
        }
        __syncthreads();
        // ---- diag[t] = sum_k r*u*k (raw tiles from staging) ----
        {
            int row = tid >> 1, half = (tid & 1) * 32;
            float s = 0.f;
#pragma unroll
            for (int j = 0; j < 32; ++j) {
                int c = half + j;
                s += bf2f(sm.r_st[row * 64 + c]) * sm.uu[c] * bf2f(sm.k_st[row * 64 + c]);
            }
            s += __shfl_down(s, 1);
            if ((tid & 1) == 0) sm.diag[row] = s;
        }
        __syncthreads();
        // ---- Phase A: a = rd @ kd^T, tril mask + diag, store bf16 ----
        {
            int m0 = wave * 16;
            // hoist this wave's diagonal values into registers (branchless mask)
            float dvs[8];
#pragma unroll
            for (int v = 0; v < 8; ++v) dvs[v] = sm.diag[m0 + hi * 8 + v];
            for (int nt = 0; nt < 8; ++nt) {
                f32x8 acc = (f32x8)0.0f;
#pragma unroll
                for (int ks = 0; ks < 2; ++ks) {
                    Frag a, bb;
                    int r = m0 + lr;
                    a.q[0] = *(const uint4*)&sm.rdw[r * 64 + ks * 32 + hi * 8];
                    a.q[1] = *(const uint4*)&sm.rdw[r * 64 + ks * 32 + 16 + hi * 8];
                    int wr = nt * 16 + lr;
                    bb.q[0] = *(const uint4*)&sm.kd[wr * 64 + ks * 32 + hi * 8];
                    bb.q[1] = *(const uint4*)&sm.kd[wr * 64 + ks * 32 + 16 + hi * 8];
                    acc = wmma_bf16(a.v, bb.v, acc);
                }
#pragma unroll
                for (int v = 0; v < 8; ++v) {
                    int m = m0 + hi * 8 + v;
                    int scol = nt * 16 + lr;
                    float val = (scol > m) ? 0.f : acc[v];   // strict tril
                    val = (scol == m) ? dvs[v] : val;        // diagonal = r*u*k
                    sm.a_bf[m * 128 + scol] = f2bf(val);
                }
            }
        }
        __syncthreads();
        // ---- rd -> rw (*= w^64), state -> stateT (bf16) ----
#pragma unroll 4
        for (int i = 0; i < 32; ++i) {
            int e = tid + i * 256;
            int col = e & 63;
            sm.rdw[e] = f2bf(bf2f(sm.rdw[e]) * __expf(64.f * sm.wl[col]));
        }
#pragma unroll 4
        for (int i = 0; i < 16; ++i) {
            int e = tid + i * 256;
            int kk = e >> 6, vv = e & 63;
            sm.stateT[vv * 64 + kk] = f2bf(sm.state[e]);
        }
        __syncthreads();
        // ---- Phase B: out = a @ v + rw @ state ----
        {
            int m0 = wave * 16;
            f32x8 acc[4];
#pragma unroll
            for (int j = 0; j < 4; ++j) acc[j] = (f32x8)0.0f;
#pragma unroll
            for (int ks = 0; ks < 4; ++ks) {
                Frag a;
                int r = m0 + lr;
                a.q[0] = *(const uint4*)&sm.a_bf[r * 128 + ks * 32 + hi * 8];
                a.q[1] = *(const uint4*)&sm.a_bf[r * 128 + ks * 32 + 16 + hi * 8];
#pragma unroll
                for (int j = 0; j < 4; ++j) {
                    Frag bb;
                    int vr = j * 16 + lr;
                    bb.q[0] = *(const uint4*)&sm.vT[vr * 128 + ks * 32 + hi * 8];
                    bb.q[1] = *(const uint4*)&sm.vT[vr * 128 + ks * 32 + 16 + hi * 8];
                    acc[j] = wmma_bf16(a.v, bb.v, acc[j]);
                }
            }
#pragma unroll
            for (int ks = 0; ks < 2; ++ks) {
                Frag a;
                int r = m0 + lr;
                a.q[0] = *(const uint4*)&sm.rdw[r * 64 + ks * 32 + hi * 8];
                a.q[1] = *(const uint4*)&sm.rdw[r * 64 + ks * 32 + 16 + hi * 8];
#pragma unroll
                for (int j = 0; j < 4; ++j) {
                    Frag bb;
                    int vr = j * 16 + lr;
                    bb.q[0] = *(const uint4*)&sm.stateT[vr * 64 + ks * 32 + hi * 8];
                    bb.q[1] = *(const uint4*)&sm.stateT[vr * 64 + ks * 32 + 16 + hi * 8];
                    acc[j] = wmma_bf16(a.v, bb.v, acc[j]);
                }
            }
#pragma unroll
            for (int j = 0; j < 4; ++j)
#pragma unroll
                for (int v = 0; v < 8; ++v) {
                    int m = m0 + hi * 8 + v;
                    att_out[gbase + (size_t)m * Csz + j * 16 + lr] = acc[j][v];
                }
        }
        __syncthreads();
        // ---- Phase C: wkv = kw^T @ v ; state = state * w^128 + wkv ----
        {
            int cm = (wave & 3) * 16, cn = (wave >> 2) * 32;
            f32x8 acc[2];
            acc[0] = (f32x8)0.0f;
            acc[1] = (f32x8)0.0f;
#pragma unroll
            for (int ks = 0; ks < 4; ++ks) {
                Frag a;
                int r = cm + lr;
                a.q[0] = *(const uint4*)&sm.kwT[r * 128 + ks * 32 + hi * 8];
                a.q[1] = *(const uint4*)&sm.kwT[r * 128 + ks * 32 + 16 + hi * 8];
#pragma unroll
                for (int j = 0; j < 2; ++j) {
                    Frag bb;
                    int vr = cn + j * 16 + lr;
                    bb.q[0] = *(const uint4*)&sm.vT[vr * 128 + ks * 32 + hi * 8];
                    bb.q[1] = *(const uint4*)&sm.vT[vr * 128 + ks * 32 + 16 + hi * 8];
                    acc[j] = wmma_bf16(a.v, bb.v, acc[j]);
                }
            }
#pragma unroll
            for (int j = 0; j < 2; ++j)
#pragma unroll
                for (int v = 0; v < 8; ++v) {
                    int kk = cm + hi * 8 + v;
                    int vv = cn + j * 16 + lr;
                    float dec = __expf(128.f * sm.wl[kk]);
                    sm.state[kk * 64 + vv] = sm.state[kk * 64 + vv] * dec + acc[j][v];
                }
        }
        __syncthreads();
    }
    for (int i = tid; i < 4096; i += 256)
        state_out[(size_t)(b * Hsz + h) * 4096 + i] = sm.state[i];
}

// ---------------------------------------------------------------------------
// K4: groupnorm (per b,l,h over 64) + silu(g) gating -> bf16
// one wave per group; wave32 shuffle reductions
// ---------------------------------------------------------------------------
__global__ void __launch_bounds__(256) gn_gate(
    const float* __restrict__ att, const unsigned short* __restrict__ G,
    const float* __restrict__ lnw, const float* __restrict__ lnb,
    unsigned short* __restrict__ ng) {
    int tid = threadIdx.x;
    int lane = tid & 31;
    int grp = blockIdx.x * 8 + (tid >> 5);
    int h = grp & (Hsz - 1);
    size_t bl = (size_t)(grp >> 5);
    size_t base = bl * Csz + (size_t)h * Ksz;
    float x0 = att[base + lane] * 0.125f;        // / HS_DIVISOR
    float x1 = att[base + 32 + lane] * 0.125f;
    float s = x0 + x1;
#pragma unroll
    for (int m = 16; m; m >>= 1) s += __shfl_xor(s, m);
    float mean = s * (1.f / 64.f);
    float d0 = x0 - mean, d1 = x1 - mean;
    float v = d0 * d0 + d1 * d1;
#pragma unroll
    for (int m = 16; m; m >>= 1) v += __shfl_xor(v, m);
    float rstd = rsqrtf(v * (1.f / 64.f) + 1e-5f);
#pragma unroll
    for (int t = 0; t < 2; ++t) {
        int c = lane + t * 32;
        float d = t ? d1 : d0;
        float xn = d * rstd * lnw[h * Ksz + c] + lnb[h * Ksz + c];
        float g = bf2f(G[base + c]);
        float sg = g / (1.f + __expf(-g));
        ng[base + c] = f2bf(xn * sg);
    }
}

// ---------------------------------------------------------------------------
// K6: x[:, -1, :] -> d_out tail
// ---------------------------------------------------------------------------
__global__ void __launch_bounds__(256) xlast(const float* __restrict__ x,
                                             float* __restrict__ out) {
    int i = blockIdx.x * 256 + threadIdx.x;  // B*C = 8192
    int b = i >> 11, c = i & (Csz - 1);
    out[i] = x[((size_t)b * Lsz + (Lsz - 1)) * Csz + c];
}

// ---------------------------------------------------------------------------
extern "C" void kernel_launch(void* const* d_in, const int* in_sizes, int n_in,
                              void* d_out, int out_size, void* d_ws, size_t ws_size,
                              hipStream_t stream) {
    (void)in_sizes; (void)n_in; (void)out_size; (void)ws_size;
    const float* x      = (const float*)d_in[0];
    const float* shift  = (const float*)d_in[1];
    const float* wkv_in = (const float*)d_in[2];
    const float* tmk    = (const float*)d_in[3];
    const float* tmv    = (const float*)d_in[4];
    const float* tmr    = (const float*)d_in[5];
    const float* tmg    = (const float*)d_in[6];
    const float* tdec   = (const float*)d_in[7];
    const float* tfaa   = (const float*)d_in[8];
    const float* Wr     = (const float*)d_in[9];
    const float* Wk     = (const float*)d_in[10];
    const float* Wv     = (const float*)d_in[11];
    const float* Wg     = (const float*)d_in[12];
    const float* Wo     = (const float*)d_in[13];
    const float* lnw    = (const float*)d_in[14];
    const float* lnb    = (const float*)d_in[15];

    char* ws = (char*)d_ws;
    const size_t WB  = (size_t)Csz * Csz * 2;  // bf16 weight: 8 MB
    const size_t XB  = (size_t)Msz * Csz * 2;  // bf16 activation: 32 MB
    size_t o = 0;
    unsigned short* Wr_bf = (unsigned short*)(ws + o); o += WB;
    unsigned short* Wk_bf = (unsigned short*)(ws + o); o += WB;
    unsigned short* Wv_bf = (unsigned short*)(ws + o); o += WB;
    unsigned short* Wg_bf = (unsigned short*)(ws + o); o += WB;
    unsigned short* Wo_bf = (unsigned short*)(ws + o); o += WB;
    unsigned short* xr_bf = (unsigned short*)(ws + o); o += XB;
    unsigned short* xk_bf = (unsigned short*)(ws + o); o += XB;
    unsigned short* xv_bf = (unsigned short*)(ws + o); o += XB;
    unsigned short* xg_bf = (unsigned short*)(ws + o); o += XB;
    unsigned short* r_bf  = (unsigned short*)(ws + o); o += XB;
    unsigned short* k_bf  = (unsigned short*)(ws + o); o += XB;
    unsigned short* v_bf  = (unsigned short*)(ws + o); o += XB;
    unsigned short* G_bf  = (unsigned short*)(ws + o); o += XB;
    float*          att   = (float*)(ws + o);          o += (size_t)Msz * Csz * 4;
    unsigned short* ng_bf = (unsigned short*)(ws + o); o += XB;

    float* out_f = (float*)d_out;
    float* out_xlast = out_f + (size_t)Bsz * Lsz * Csz;           // 16777216
    float* out_state = out_xlast + (size_t)Bsz * Csz;             // + 8192

    // weight conversion
    int wcb = (Csz * Csz) / 256;
    cvt_w<<<wcb, 256, 0, stream>>>(Wr, Wr_bf);
    cvt_w<<<wcb, 256, 0, stream>>>(Wk, Wk_bf);
    cvt_w<<<wcb, 256, 0, stream>>>(Wv, Wv_bf);
    cvt_w<<<wcb, 256, 0, stream>>>(Wg, Wg_bf);
    cvt_w<<<wcb, 256, 0, stream>>>(Wo, Wo_bf);

    // token shift + lerp
    prep_lerp<<<(Bsz * Lsz * Csz) / 256, 256, 0, stream>>>(
        x, shift, tmk, tmv, tmr, tmg, xk_bf, xv_bf, xr_bf, xg_bf);

    // projections (NT bf16 GEMMs, async-LDS pipelined)
    dim3 gg(Csz / 128, Msz / 128);
    gemm_nt<0><<<gg, 256, 0, stream>>>(xr_bf, Wr_bf, r_bf, Msz, Csz, Csz);
    gemm_nt<0><<<gg, 256, 0, stream>>>(xk_bf, Wk_bf, k_bf, Msz, Csz, Csz);
    gemm_nt<0><<<gg, 256, 0, stream>>>(xv_bf, Wv_bf, v_bf, Msz, Csz, Csz);
    gemm_nt<0><<<gg, 256, 0, stream>>>(xg_bf, Wg_bf, G_bf, Msz, Csz, Csz);

    // chunked wkv attention (one WG per (b,h); sequential chunk scan, TDM loads)
    rwkv_att<<<Bsz * Hsz, 256, sizeof(SmemK3), stream>>>(
        r_bf, k_bf, v_bf, tdec, tfaa, wkv_in, att, out_state);

    // groupnorm + silu gate
    gn_gate<<<(Bsz * Lsz * Hsz) / 8, 256, 0, stream>>>(att, G_bf, lnw, lnb, ng_bf);

    // output projection (f32 out straight to d_out)
    gemm_nt<1><<<gg, 256, 0, stream>>>(ng_bf, Wo_bf, (void*)out_f, Msz, Csz, Csz);

    // x[:, -1]
    xlast<<<(Bsz * Csz) / 256, 256, 0, stream>>>(x, out_xlast);
}
